// KGAT_45294725104259
// MI455X (gfx1250) — compile-verified, matrix operand order checked by reading, not run on hardware
//
#include <hip/hip_runtime.h>
#include <hip/hip_bf16.h>

typedef __attribute__((ext_vector_type(16))) _Float16 v16h;
typedef __attribute__((ext_vector_type(8)))  float    v8f;

static __device__ __forceinline__ v8f wmma_f16(v16h a, v16h b, v8f c) {
    // 8 args: (neg_a, A, neg_b, B, c_mod, C, reuse_a, reuse_b)
    return __builtin_amdgcn_wmma_f32_16x16x32_f16(false, a, false, b, (short)0, c, false, false);
}

#define LRELU(x) ((x) > 0.0f ? (x) : 0.01f * (x))

// ---------------------------------------------------------------------------
// out[:, 0:64] = ue_emb  (first block of the concat), float4 per thread
// ---------------------------------------------------------------------------
__global__ void kgat_copy_emb(const float* __restrict__ src, float* __restrict__ dst,
                              int out_stride, long long nchunks /* N*16 */) {
    long long gid = (long long)blockIdx.x * blockDim.x + threadIdx.x;
    if (gid >= nchunks) return;
    long long r = gid >> 4;
    int c = (int)(gid & 15) << 2;
    *(float4*)(dst + r * out_stride + c) = *(const float4*)(src + r * 64 + c);
}

// ---------------------------------------------------------------------------
// Pack W [DIN, DOUT] (f32, row-major) into per-lane v16h WMMA-B operand order:
//   P[((nt*NKT + kt)*32 + lane)*16 + i] = (f16) W[K*DOUT + col]
//   K = kt*32 + (i&7) + ((i>>3)<<4) + ((lane>>4)<<3),  col = nt*16 + (lane&15)
// ---------------------------------------------------------------------------
template <int DIN, int DOUT>
__global__ void kgat_pack_w(const float* __restrict__ W, _Float16* __restrict__ P) {
    constexpr int NKT = DIN / 32;
    int t = blockIdx.x * blockDim.x + threadIdx.x;
    if (t >= DIN * DOUT) return;
    int i    = t & 15;
    int lane = (t >> 4) & 31;
    int tile = t >> 9;                 // nt*NKT + kt
    int kt   = tile % NKT;
    int nt   = tile / NKT;
    int hi   = lane >> 4;
    int K    = (kt << 5) + (i & 7) + ((i >> 3) << 4) + (hi << 3);
    int col  = (nt << 4) + (lane & 15);
    P[t] = (_Float16)W[K * DOUT + col];
}

// ---------------------------------------------------------------------------
// side[dst, :] += vals[e] * ego[src, :]   (DIN/4 lanes per edge, float4 each)
// ---------------------------------------------------------------------------
template <int DIN>
__global__ void kgat_spmm(const float* __restrict__ ego, int ego_stride,
                          const float* __restrict__ vals,
                          const int* __restrict__ rows,
                          const int* __restrict__ cols,
                          float* __restrict__ side, long long E) {
    constexpr int CSHIFT = (DIN == 64) ? 4 : 3;    // log2(DIN/4)
    long long gid = (long long)blockIdx.x * blockDim.x + threadIdx.x;
    long long total = E << CSHIFT;
    if (gid >= total) return;
    long long e = gid >> CSHIFT;
    int c = (int)(gid & ((1 << CSHIFT) - 1));

    int src = cols[e];
    int dst = rows[e];
    float v = vals[e];

    const float4 x = *(const float4*)(ego + (long long)src * ego_stride + c * 4);
    float* sp = side + (long long)dst * DIN + c * 4;
    unsafeAtomicAdd(sp + 0, v * x.x);
    unsafeAtomicAdd(sp + 1, v * x.y);
    unsafeAtomicAdd(sp + 2, v * x.z);
    unsafeAtomicAdd(sp + 3, v * x.w);
}

// ---------------------------------------------------------------------------
// Bi-Interaction aggregator, fully specialized, WMMA f16->f32, f32 epilogue:
//   sum = lrelu((ego+side)@W1 + b1); bi = lrelu((ego*side)@W2 + b2)
//   out = l2_normalize(sum + bi)   (row stride out_stride)
// One wave per 16-row tile; 8 waves/block; wave-private LDS slab; no barriers.
// ---------------------------------------------------------------------------
template <int DIN, int DOUT>
__global__ void __launch_bounds__(256)
kgat_bi_agg(const float* __restrict__ ego, int ego_stride,
            const float* __restrict__ side,
            const _Float16* __restrict__ P1, const float* __restrict__ b1,
            const _Float16* __restrict__ P2, const float* __restrict__ b2,
            float* __restrict__ out, int out_stride, int N) {
    constexpr int NKT = DIN / 32;                  // 1 or 2 K-tiles
    constexpr int NT  = DOUT / 16;                 // 4, 2 or 1 N-tiles
    constexpr int CPSH = (DOUT == 64) ? 4 : ((DOUT == 32) ? 3 : 2);

    __shared__ float lds[8][16 * 64 + 16];

    const int wave = threadIdx.x >> 5;
    const int lane = threadIdx.x & 31;
    const int rowbase = (blockIdx.x * 8 + wave) * 16;
    if (rowbase >= N) return;                      // 16 | N: waves all-or-nothing

    const int m  = lane & 15;
    const int hi = lane >> 4;
    const long long arow = rowbase + m;

    // --- A operands for both branches, vectorized float4 loads -------------
    v16h aS[NKT], aB[NKT];
    const float* erow = ego  + arow * ego_stride;
    const float* srow = side + arow * DIN;
#pragma unroll
    for (int kt = 0; kt < NKT; ++kt) {
        const int K0 = (kt << 5) + (hi << 3);
        float4 e0 = *(const float4*)(erow + K0);
        float4 e1 = *(const float4*)(erow + K0 + 4);
        float4 e2 = *(const float4*)(erow + K0 + 16);
        float4 e3 = *(const float4*)(erow + K0 + 20);
        float4 s0 = *(const float4*)(srow + K0);
        float4 s1 = *(const float4*)(srow + K0 + 4);
        float4 s2 = *(const float4*)(srow + K0 + 16);
        float4 s3 = *(const float4*)(srow + K0 + 20);
        float ee[16] = {e0.x, e0.y, e0.z, e0.w, e1.x, e1.y, e1.z, e1.w,
                        e2.x, e2.y, e2.z, e2.w, e3.x, e3.y, e3.z, e3.w};
        float ss[16] = {s0.x, s0.y, s0.z, s0.w, s1.x, s1.y, s1.z, s1.w,
                        s2.x, s2.y, s2.z, s2.w, s3.x, s3.y, s3.z, s3.w};
#pragma unroll
        for (int i = 0; i < 16; ++i) {
            aS[kt][i] = (_Float16)(ee[i] + ss[i]);
            aB[kt][i] = (_Float16)(ee[i] * ss[i]);
        }
    }

    // --- GEMM over N tiles, pre-packed B operands (one v16h load each) -----
    const v16h* Bp1 = (const v16h*)P1;
    const v16h* Bp2 = (const v16h*)P2;
    float* hrow  = lds[wave];
    float* scale = hrow + 16 * 64;

#pragma unroll
    for (int nt = 0; nt < NT; ++nt) {
        v8f acc1 = {};
        v8f acc2 = {};
        const int col = (nt << 4) + m;
#pragma unroll
        for (int kt = 0; kt < NKT; ++kt) {
            const int tidx = (nt * NKT + kt) * 32 + lane;
            v16h bv1 = Bp1[tidx];
            v16h bv2 = Bp2[tidx];
            acc1 = wmma_f16(aS[kt], bv1, acc1);
            acc2 = wmma_f16(aB[kt], bv2, acc2);
        }
        const float bb1 = b1[col];
        const float bb2 = b2[col];
#pragma unroll
        for (int r = 0; r < 8; ++r) {
            float v1 = acc1[r] + bb1; v1 = LRELU(v1);
            float v2 = acc2[r] + bb2; v2 = LRELU(v2);
            hrow[(r + (hi << 3)) * DOUT + col] = v1 + v2;
        }
    }

    asm volatile("s_wait_dscnt 0" ::: "memory");   // tile fully in LDS

    if (lane < 16) {                               // one lane per row: L2 norm
        float ssum = 0.0f;
#pragma unroll
        for (int c = 0; c < DOUT; c += 4) {
            float4 t = *(const float4*)(hrow + lane * DOUT + c);
            ssum += t.x * t.x + t.y * t.y + t.z * t.z + t.w * t.w;
        }
        scale[lane] = 1.0f / fmaxf(sqrtf(ssum), 1e-12f);
    }

    asm volatile("s_wait_dscnt 0" ::: "memory");   // scales visible wave-wide

#pragma unroll
    for (int it = 0; it < DOUT / 8; ++it) {        // 16*DOUT/4 chunks / 32 lanes
        int idx = it * 32 + lane;
        int r = idx >> CPSH;
        int c = (idx & ((1 << CPSH) - 1)) << 2;
        float4 h = *(const float4*)(hrow + r * DOUT + c);
        float sc = scale[r];
        float4 o = {h.x * sc, h.y * sc, h.z * sc, h.w * sc};
        *(float4*)(out + (long long)(rowbase + r) * out_stride + c) = o;
    }
}

// ---------------------------------------------------------------------------
extern "C" void kernel_launch(void* const* d_in, const int* in_sizes, int n_in,
                              void* d_out, int out_size, void* d_ws, size_t ws_size,
                              hipStream_t stream) {
    const float* ue   = (const float*)d_in[0];
    const float* vals = (const float*)d_in[1];
    const int*   rows = (const int*)d_in[2];
    const int*   cols = (const int*)d_in[3];

    const float* W1[3] = { (const float*)d_in[4],  (const float*)d_in[8],  (const float*)d_in[12] };
    const float* B1[3] = { (const float*)d_in[5],  (const float*)d_in[9],  (const float*)d_in[13] };
    const float* W2[3] = { (const float*)d_in[6],  (const float*)d_in[10], (const float*)d_in[14] };
    const float* B2[3] = { (const float*)d_in[7],  (const float*)d_in[11], (const float*)d_in[15] };

    const int  N = in_sizes[0] / 64;
    const long long E = in_sizes[1];
    const int  OS = 176;                         // 64 + 64 + 32 + 16

    float* out  = (float*)d_out;
    float* side = (float*)d_ws;                  // N*64 floats (51.2 MB)

    // Packed f16 WMMA-B weights live right after the side buffer (32B aligned).
    _Float16* pbase = (_Float16*)((char*)d_ws + (size_t)N * 64 * sizeof(float));
    _Float16* P1_0 = pbase;                      // 64*64
    _Float16* P2_0 = P1_0 + 64 * 64;
    _Float16* P1_1 = P2_0 + 64 * 64;             // 64*32
    _Float16* P2_1 = P1_1 + 64 * 32;
    _Float16* P1_2 = P2_1 + 64 * 32;             // 32*16
    _Float16* P2_2 = P1_2 + 32 * 16;

    kgat_pack_w<64, 64><<<(64 * 64 + 255) / 256, 256, 0, stream>>>(W1[0], P1_0);
    kgat_pack_w<64, 64><<<(64 * 64 + 255) / 256, 256, 0, stream>>>(W2[0], P2_0);
    kgat_pack_w<64, 32><<<(64 * 32 + 255) / 256, 256, 0, stream>>>(W1[1], P1_1);
    kgat_pack_w<64, 32><<<(64 * 32 + 255) / 256, 256, 0, stream>>>(W2[1], P2_1);
    kgat_pack_w<32, 16><<<(32 * 16 + 255) / 256, 256, 0, stream>>>(W1[2], P1_2);
    kgat_pack_w<32, 16><<<(32 * 16 + 255) / 256, 256, 0, stream>>>(W2[2], P2_2);

    // out[:, 0:64] = ue_emb
    {
        long long chunks = (long long)N * 16;
        kgat_copy_emb<<<(int)((chunks + 255) / 256), 256, 0, stream>>>(ue, out, OS, chunks);
    }

    const int gblocks = (N + 127) / 128;         // 8 waves * 16 rows per block

    // ---- Layer 0: din=64, dout=64, cols [64,128) ---------------------------
    hipMemsetAsync(side, 0, (size_t)N * 64 * sizeof(float), stream);
    kgat_spmm<64><<<(int)((E * 16 + 255) / 256), 256, 0, stream>>>(
        ue, 64, vals, rows, cols, side, E);
    kgat_bi_agg<64, 64><<<gblocks, 256, 0, stream>>>(
        ue, 64, side, P1_0, B1[0], P2_0, B2[0], out + 64, OS, N);

    // ---- Layer 1: din=64, dout=32, cols [128,160) --------------------------
    hipMemsetAsync(side, 0, (size_t)N * 64 * sizeof(float), stream);
    kgat_spmm<64><<<(int)((E * 16 + 255) / 256), 256, 0, stream>>>(
        out + 64, OS, vals, rows, cols, side, E);
    kgat_bi_agg<64, 32><<<gblocks, 256, 0, stream>>>(
        out + 64, OS, side, P1_1, B1[1], P2_1, B2[1], out + 128, OS, N);

    // ---- Layer 2: din=32, dout=16, cols [160,176) --------------------------
    hipMemsetAsync(side, 0, (size_t)N * 32 * sizeof(float), stream);
    kgat_spmm<32><<<(int)((E * 8 + 255) / 256), 256, 0, stream>>>(
        out + 128, OS, vals, rows, cols, side, E);
    kgat_bi_agg<32, 16><<<gblocks, 256, 0, stream>>>(
        out + 128, OS, side, P1_2, B1[2], P2_2, B2[2], out + 160, OS, N);
}
